// FLAVORplus_36850819399684
// MI455X (gfx1250) — compile-verified, matrix operand order checked by reading, not run on hardware
//
#include <hip/hip_runtime.h>
#include <hip/hip_bf16.h>

// Problem constants (from reference)
#define Bc 4
#define Hc 4
#define Lc 8192
#define Dc 128      // head dim (also output cols)
#define Mc 64       // projection dim
#define Fc 128      // feature dim = 2*Mc
#define NP 144      // padded C columns: 128 V cols + 1 ones col + 15 zero pad
#define RL 128      // L rows per chunk
#define CH 4        // chunks accumulated per phase-1 block (cuts atomics 4x)
#define LP 136      // LDS row stride (halves) to dodge bank conflicts
#define EPSf 1.0e-4f
// combined scale: exp(-0.5*ss)/sqrt(2)/sqrt(M) with M=64 -> /(8*sqrt(2))
#define GSCL 0.08838834764831845f

typedef __attribute__((ext_vector_type(16))) _Float16 v16h;
typedef __attribute__((ext_vector_type(8)))  float    v8f;
typedef __attribute__((ext_vector_type(4)))  float    f4;

// Load a 16x32 (rows x K) f16 fragment in the CDNA5 WMMA A/B register layout.
// 'mat' is row-major with stride LP halves; for B operands the matrix is the
// TRANSPOSE (lane indexes the B column). Pattern per ISA 7.12.2:
//   lanes 0-15: VGPR0-3 K=0..7, VGPR4-7 K=16..23 ; lanes 16-31: +8
__device__ __forceinline__ v16h load_frag(const _Float16* mat, int row0, int k0, int lane) {
  int r   = row0 + (lane & 15);
  int grp = (lane >> 4) & 1;
  const _Float16* p = mat + r * LP + k0 + grp * 8;
  union { v16h v; _Float16 h[16]; } u;
#pragma unroll
  for (int j = 0; j < 4; ++j) { u.h[2*j]   = p[2*j];      u.h[2*j+1] = p[2*j+1]; }
  const _Float16* q = p + 16;
#pragma unroll
  for (int j = 0; j < 4; ++j) { u.h[8+2*j] = q[2*j];      u.h[9+2*j] = q[2*j+1]; }
  return u.v;
}

__device__ __forceinline__ v8f wmma_f16(v16h a, v16h b, v8f c) {
  return __builtin_amdgcn_wmma_f32_16x16x32_f16(false, a, false, b, (short)0, c, false, false);
}

// 32-row x (NTILES*16)-col x K=128 GEMM strip from LDS. Each B fragment feeds
// two WMMAs (A-pair reuse) -> 2A+NTILES B loads per K-step for 2*NTILES WMMAs,
// cutting LDS fragment traffic ~35% vs a 1xN partition. B is prefetched one
// tile ahead of its consuming WMMA pair.
template <int NTILES>
__device__ __forceinline__ void gemm_2xN(const _Float16* A, const _Float16* B,
                                         int m0, int nbase, int lane,
                                         v8f (*acc)[NTILES]) {
#pragma unroll
  for (int kt = 0; kt < 4; ++kt) {
    v16h a0 = load_frag(A, m0,      kt * 32, lane);
    v16h a1 = load_frag(A, m0 + 16, kt * 32, lane);
    v16h b  = load_frag(B, nbase * 16, kt * 32, lane);
#pragma unroll
    for (int j = 0; j < NTILES; ++j) {
      v16h bn = b;
      if (j + 1 < NTILES) bn = load_frag(B, (nbase + j + 1) * 16, kt * 32, lane);
      acc[0][j] = wmma_f16(a0, b, acc[0][j]);
      acc[1][j] = wmma_f16(a1, b, acc[1][j]);
      b = bn;
    }
  }
}

__global__ void flavor_zero_buf1(float* __restrict__ ws, int n) {
  int i = blockIdx.x * 256 + threadIdx.x;
  if (i < n) ws[i] = 0.0f;
}

// ---------------- Phase 1: Buf1[bh] += phi(K)^T @ [V | 1] -------------------
__global__ __launch_bounds__(256) void flavor_phase1(
    const float* __restrict__ Kg, const float* __restrict__ Vg,
    const float* __restrict__ wg, float* __restrict__ buf1g) {
  __shared__ _Float16 sA[RL][LP];   // K chunk f16; later aliased as Kp^T [feat][row]
  __shared__ _Float16 sW[Mc][LP];   // w^T : [m][d]
  __shared__ _Float16 sCt[NP][LP];  // C^T : [col][row]
  __shared__ float    sG[RL];       // per-row combined gate scale
  __shared__ float    sPart[256];

  const int tid  = threadIdx.x;
  const int lane = tid & 31;
  const int wv   = tid >> 5;            // 8 waves
  const int mh   = wv & 3;              // M-tile pair: rows 32*mh .. +31
  const int nh   = wv >> 2;             // N strip half
  const int bh   = blockIdx.y;
  const int h    = bh & (Hc - 1);
  const int nidx = lane & 15, grp = lane >> 4;
  const int r    = tid >> 1, seg = tid & 1;

  // Load w^T once (column n of w into LDS row n); small, stays hot in L2
#pragma unroll
  for (int i = 0; i < 32; ++i) {
    int idx = tid + 256 * i;            // 0..8191
    int m = idx & 63, d = idx >> 6;
    sW[m][d] = (_Float16)wg[((size_t)h * Dc + d) * Mc + m];
  }

  // Buf1 accumulators, live across chunks: 2 Mtiles x 5 col tiles
  // (nh=0: tiles 0..4, nh=1: tiles 4..8; tile 4 duplicated, skipped in epilogue)
  v8f bacc[2][5];
#pragma unroll
  for (int i = 0; i < 2; ++i)
#pragma unroll
    for (int j = 0; j < 5; ++j) bacc[i][j] = {};

  for (int ch = 0; ch < CH; ++ch) {
    const int l0 = (blockIdx.x * CH + ch) * RL;

    // Stage 1: load K chunk (coalesced, non-temporal), f32->f16, row sumsq
    {
      const f4* src = (const f4*)(Kg + (((size_t)bh * Lc + l0 + r) * Dc) + seg * 64);
      float ss = 0.0f;
#pragma unroll
      for (int i = 0; i < 16; ++i) {
        f4 v = __builtin_nontemporal_load(src + i);
        ss += v.x*v.x + v.y*v.y + v.z*v.z + v.w*v.w;
        int c = seg * 64 + i * 4;
        sA[r][c+0] = (_Float16)v.x; sA[r][c+1] = (_Float16)v.y;
        sA[r][c+2] = (_Float16)v.z; sA[r][c+3] = (_Float16)v.w;
      }
      sPart[tid] = ss;
    }
    __syncthreads();
    if (tid < RL) sG[tid] = __expf(-0.5f * (sPart[2*tid] + sPart[2*tid+1])) * GSCL;
    __syncthreads();

    // Stage 2: o = K @ w  (wave owns rows 32mh..+31, cols 32nh..+31)
    v8f oacc[2][2];
#pragma unroll
    for (int i = 0; i < 2; ++i)
#pragma unroll
      for (int j = 0; j < 2; ++j) oacc[i][j] = {};
    gemm_2xN<2>(&sA[0][0], &sW[0][0], mh * 32, nh * 2, lane, oacc);
    __syncthreads();   // GEMM1 LDS reads done before overwriting sA with Kp^T

    // Stage 3: features Kp^T[feat][row] = (exp(+-o)+eps)*g   (aliases sA)
    {
      _Float16 (*sPt)[LP] = sA;
#pragma unroll
      for (int i = 0; i < 2; ++i) {
#pragma unroll
        for (int j = 0; j < 2; ++j) {
#pragma unroll
          for (int rr = 0; rr < 8; ++rr) {
            int row = (2 * mh + i) * 16 + rr + 8 * grp;
            int col = (2 * nh + j) * 16 + nidx;
            float o = oacc[i][j][rr];
            float s = sG[row];
            sPt[col][row]      = (_Float16)((__expf(o)  + EPSf) * s);
            sPt[col + Mc][row] = (_Float16)((__expf(-o) + EPSf) * s);
          }
        }
      }
    }
    // Stage 4: C^T = [V | 1 | 0-pad]^T
    {
      const f4* src = (const f4*)(Vg + (((size_t)bh * Lc + l0 + r) * Dc) + seg * 64);
#pragma unroll
      for (int i = 0; i < 16; ++i) {
        f4 v = __builtin_nontemporal_load(src + i);
        int c = seg * 64 + i * 4;
        sCt[c+0][r] = (_Float16)v.x; sCt[c+1][r] = (_Float16)v.y;
        sCt[c+2][r] = (_Float16)v.z; sCt[c+3][r] = (_Float16)v.w;
      }
      if (tid < RL) {
        sCt[Dc][tid] = (_Float16)1.0f;
#pragma unroll
        for (int c = Dc + 1; c < NP; ++c) sCt[c][tid] = (_Float16)0.0f;
      }
    }
    __syncthreads();

    // Stage 5: Buf1 += Kp^T @ C  (feature rows 32mh..+31, col tiles nh*4..+4)
    gemm_2xN<5>(&sA[0][0], &sCt[0][0], mh * 32, nh * 4, lane, bacc);
    __syncthreads();   // GEMM2 reads done before next chunk overwrites sA/sCt
  }

  // One atomic epilogue per block (CH chunks folded in registers).
  // nh=1 skips j==0 (tile 4 was duplicated across the two N strips).
  {
    float* buf1 = buf1g + (size_t)bh * Fc * NP;
#pragma unroll
    for (int i = 0; i < 2; ++i) {
#pragma unroll
      for (int j = 0; j < 5; ++j) {
        if (nh == 1 && j == 0) continue;
#pragma unroll
        for (int rr = 0; rr < 8; ++rr) {
          int f = (2 * mh + i) * 16 + rr + 8 * grp;
          int c = (nh * 4 + j) * 16 + nidx;
          __hip_atomic_fetch_add(&buf1[f * NP + c], bacc[i][j][rr],
                                 __ATOMIC_RELAXED, __HIP_MEMORY_SCOPE_AGENT);
        }
      }
    }
  }
}

// ---------------- Phase 2: out = (phi(Q) @ Buf1) num/den --------------------
__global__ __launch_bounds__(256) void flavor_phase2(
    const float* __restrict__ Qg, const float* __restrict__ wg,
    const float* __restrict__ buf1g, float* __restrict__ outg) {
  __shared__ _Float16 sA[RL][LP];    // Q chunk f16; later aliased as Qp [row][feat]
  __shared__ _Float16 sW[Mc][LP];    // w^T
  __shared__ _Float16 sBt[NP][LP];   // Buf1^T : [col][feat]
  __shared__ float    sG[RL];
  __shared__ float    sPart[256];
  __shared__ float    sDen[RL];

  const int tid  = threadIdx.x;
  const int lane = tid & 31;
  const int wv   = tid >> 5;
  const int mh   = wv & 3;
  const int nh   = wv >> 2;
  const int l0   = blockIdx.x * RL;
  const int bh   = blockIdx.y;
  const int h    = bh & (Hc - 1);
  const int nidx = lane & 15, grp = lane >> 4;

  // Stage 1: load Q chunk (non-temporal) + row sumsq
  {
    const int r = tid >> 1, seg = tid & 1;
    const f4* src = (const f4*)(Qg + (((size_t)bh * Lc + l0 + r) * Dc) + seg * 64);
    float ss = 0.0f;
#pragma unroll
    for (int i = 0; i < 16; ++i) {
      f4 v = __builtin_nontemporal_load(src + i);
      ss += v.x*v.x + v.y*v.y + v.z*v.z + v.w*v.w;
      int c = seg * 64 + i * 4;
      sA[r][c+0] = (_Float16)v.x; sA[r][c+1] = (_Float16)v.y;
      sA[r][c+2] = (_Float16)v.z; sA[r][c+3] = (_Float16)v.w;
    }
    sPart[tid] = ss;
  }
  // Stage 2: w^T
#pragma unroll
  for (int i = 0; i < 32; ++i) {
    int idx = tid + 256 * i;
    int m = idx & 63, d = idx >> 6;
    sW[m][d] = (_Float16)wg[((size_t)h * Dc + d) * Mc + m];
  }
  // Stage 2b: Buf1^T f32->f16  (128*144 = 18432 = 256*72); hot in L2
  {
    const float* b1 = buf1g + (size_t)bh * Fc * NP;
#pragma unroll
    for (int i = 0; i < 72; ++i) {
      int idx = tid + 256 * i;
      int f = idx / NP;
      int c = idx - f * NP;
      sBt[c][f] = (_Float16)b1[idx];
    }
  }
  __syncthreads();
  if (tid < RL) sG[tid] = __expf(-0.5f * (sPart[2*tid] + sPart[2*tid+1])) * GSCL;
  __syncthreads();

  // Stage 3: o = Q @ w
  v8f oacc[2][2];
#pragma unroll
  for (int i = 0; i < 2; ++i)
#pragma unroll
    for (int j = 0; j < 2; ++j) oacc[i][j] = {};
  gemm_2xN<2>(&sA[0][0], &sW[0][0], mh * 32, nh * 2, lane, oacc);
  __syncthreads();

  // Stage 4: Qp row-major [row][feat] (aliases sA)
  {
#pragma unroll
    for (int i = 0; i < 2; ++i) {
#pragma unroll
      for (int j = 0; j < 2; ++j) {
#pragma unroll
        for (int rr = 0; rr < 8; ++rr) {
          int row = (2 * mh + i) * 16 + rr + 8 * grp;
          int col = (2 * nh + j) * 16 + nidx;
          float o = oacc[i][j][rr];
          float s = sG[row];
          sA[row][col]      = (_Float16)((__expf(o)  + EPSf) * s);
          sA[row][col + Mc] = (_Float16)((__expf(-o) + EPSf) * s);
        }
      }
    }
  }
  __syncthreads();

  // Stage 5: Buf2 = Qp @ Buf1  (rows 32mh..+31, col tiles nh*4..+4)
  v8f bacc[2][5];
#pragma unroll
  for (int i = 0; i < 2; ++i)
#pragma unroll
    for (int j = 0; j < 5; ++j) bacc[i][j] = {};
  gemm_2xN<5>(&sA[0][0], &sBt[0][0], mh * 32, nh * 4, lane, bacc);

  // Stage 6: denominator = col tile 8 (nh=1, j=4, n==0); broadcast via LDS,
  // then divide and store (nh=1 skips duplicated tile 4 and the den tile).
  {
    if (nh == 1 && nidx == 0) {
#pragma unroll
      for (int i = 0; i < 2; ++i)
#pragma unroll
        for (int rr = 0; rr < 8; ++rr)
          sDen[(2 * mh + i) * 16 + rr + 8 * grp] = bacc[i][4][rr];
    }
    __syncthreads();
    float* outp = outg + ((size_t)bh * Lc + l0) * Dc;
#pragma unroll
    for (int i = 0; i < 2; ++i) {
#pragma unroll
      for (int j = 0; j < 5; ++j) {
        if (nh == 1 && (j == 0 || j == 4)) continue;   // dup tile 4 / den tile
#pragma unroll
        for (int rr = 0; rr < 8; ++rr) {
          int row = (2 * mh + i) * 16 + rr + 8 * grp;
          float den = sDen[row];
          __builtin_nontemporal_store(bacc[i][j][rr] / den,
              &outp[(size_t)row * Dc + (nh * 4 + j) * 16 + nidx]);
        }
      }
    }
  }
}

extern "C" void kernel_launch(void* const* d_in, const int* in_sizes, int n_in,
                              void* d_out, int out_size, void* d_ws, size_t ws_size,
                              hipStream_t stream) {
  (void)in_sizes; (void)n_in; (void)out_size; (void)ws_size;
  const float* Q = (const float*)d_in[0];
  const float* K = (const float*)d_in[1];
  const float* V = (const float*)d_in[2];
  const float* w = (const float*)d_in[3];
  float* out = (float*)d_out;
  float* ws  = (float*)d_ws;

  const int buf1_elems = Bc * Hc * Fc * NP;   // 16*128*144 = 294912 f32
  flavor_zero_buf1<<<(buf1_elems + 255) / 256, 256, 0, stream>>>(ws, buf1_elems);

  dim3 grid1(Lc / (RL * CH), Bc * Hc);        // 16 x 16 = 256 blocks, 8 waves each
  flavor_phase1<<<grid1, 256, 0, stream>>>(K, V, w, ws);

  dim3 grid2(Lc / RL, Bc * Hc);               // 64 x 16 = 1024 blocks
  flavor_phase2<<<grid2, 256, 0, stream>>>(Q, w, ws, out);
}
